// MultiHeadAttentionBiGruCRFNER_4071628997375
// MI455X (gfx1250) — compile-verified, matrix operand order checked by reading, not run on hardware
//
#include <hip/hip_runtime.h>

// ---------------------------------------------------------------------------
// MHA + 2-layer BiGRU + CRF for MI455X (gfx1250, wave32, WMMA bf16 16x16x32).
// All GEMMs run on v_wmma_f32_16x16x32_bf16. Attention is flash-style
// (streamed K/V, online softmax). GRU recurrence is a persistent 32-wave
// workgroup per direction with h in LDS and Whh streamed from L2.
// GEMM tile staging uses GLOBAL_LOAD_ASYNC_TO_LDS_B128 (ASYNCcnt) when the
// toolchain exposes the builtin; otherwise falls back to synchronous staging.
// ---------------------------------------------------------------------------

typedef unsigned short u16;
typedef __attribute__((ext_vector_type(16))) __bf16 v16bf;
typedef __attribute__((ext_vector_type(8)))  float  v8f;
typedef __attribute__((ext_vector_type(4)))  int    v4i;

#define B_  32
#define S_  512
#define E_  768
#define H_  768
#define C_  9
#define NH_ 8
#define BS_ (B_ * S_)
#define G3_ (3 * H_)   // 2304

#if __has_builtin(__builtin_amdgcn_global_load_async_to_lds_b128) && \
    __has_builtin(__builtin_amdgcn_s_wait_asynccnt)
#define ASYNC_LDS 1
// Builtin takes generic v4i* params (per hipcc diagnostic); C-style cast
// drops const on the global source.
#define VPTR(p) ((v4i*)(void*)(p))
#else
#define ASYNC_LDS 0
#endif

// ---- bf16 helpers ---------------------------------------------------------
__device__ __forceinline__ u16 f2bf(float f) {
  union { float f; unsigned u; } v; v.f = f;
  unsigned r = v.u + 0x7fffu + ((v.u >> 16) & 1u);   // RNE
  return (u16)(r >> 16);
}
__device__ __forceinline__ float bf2f(u16 h) {
  union { unsigned u; float f; } v; v.u = ((unsigned)h) << 16;
  return v.f;
}
__device__ __forceinline__ __bf16 asbf(u16 x) {
  union { u16 u; __bf16 b; } c; c.u = x; return c.b;
}

// K offset of fragment VGPR i (documented 16-bit A/B layout):
// lanes 0-15 (hi=0): V0..3 -> K={0,2,4,6}, V4..7 -> K={16,18,20,22}; hi adds 8.
__device__ __forceinline__ int fragK(int i, int hi) {
  return ((i < 4) ? 2 * i : 16 + 2 * (i - 4)) + 8 * hi;
}

// Load a 16x32 bf16 fragment where element (lane&15, k) = base[row*ld + k0+k].
// Serves A-fragments (row = M) and W-style B-fragments (row = N of W[N][K]).
// The 8 VGPR pairs are two contiguous 16B runs -> lowers to ds_load_b128 x2.
__device__ __forceinline__ v16bf load_frag_row(const u16* base, int row, int ld,
                                               int k0, int hi) {
  v16bf a;
#pragma unroll
  for (int i = 0; i < 8; ++i) {
    int k = k0 + fragK(i, hi);
    a[2 * i]     = asbf(base[row * ld + k]);
    a[2 * i + 1] = asbf(base[row * ld + k + 1]);
  }
  return a;
}

__device__ __forceinline__ v8f wmma_bf16(v16bf a, v16bf b, v8f c) {
  return __builtin_amdgcn_wmma_f32_16x16x32_bf16(false, a, false, b,
                                                 (short)0, c, false, false);
}

// ---- f32 -> bf16 cast -----------------------------------------------------
__global__ void cast_f32_bf16(const float* __restrict__ in, u16* __restrict__ out,
                              size_t n) {
  size_t i = (size_t)blockIdx.x * 256 + threadIdx.x;
  if (i < n) out[i] = f2bf(in[i]);
}

// ---- generic WMMA GEMM: out[M][N] = A[M][K](bf16) @ W[N][K]^T + bias ------
// 8 waves (256 thr) as 2x4 grid, each wave 32x32 -> 64x128 macrotile.
// All launches here use M%64==0, N%128==0, K%32==0 (exact tiles).
__global__ __launch_bounds__(256)
void wmma_gemm(const u16* __restrict__ A, const u16* __restrict__ W,
               const float* __restrict__ bias, float* __restrict__ outF,
               u16* __restrict__ outB, int M, int N, int K) {
  __shared__ u16 As[64][48];    // 96B rows: 16B-aligned b128 LDS targets
  __shared__ u16 Bs[128][48];
  const int tile_n = blockIdx.x * 128;
  const int tile_m = blockIdx.y * 64;
  const int tid = threadIdx.x, lane = tid & 31, wave = tid >> 5;
  const int wm = wave >> 2, wn = wave & 3, hi = lane >> 4, ln = lane & 15;

  v8f acc[2][2] = {};
  for (int k0 = 0; k0 < K; k0 += 32) {
    __syncthreads();
    if (k0 + 32 < K) {  // prefetch next K-tile toward L2
      __builtin_prefetch(&A[(size_t)(tile_m + (tid & 63)) * K + k0 + 32], 0, 1);
      __builtin_prefetch(&W[(size_t)(tile_n + (tid & 127)) * K + k0 + 32], 0, 1);
    }
#if ASYNC_LDS
    {
      // A tile: 64x32 u16 = 256 b128 transfers, 1/thread.
      int ra = tid >> 2, ca = (tid & 3) * 8;
      __builtin_amdgcn_global_load_async_to_lds_b128(
          VPTR(&A[(size_t)(tile_m + ra) * K + k0 + ca]), VPTR(&As[ra][ca]), 0, 0);
      // B tile: 128x32 u16 = 512 b128 transfers, 2/thread.
#pragma unroll
      for (int j = 0; j < 2; ++j) {
        int idx = tid * 2 + j;
        int rb = idx >> 2, cb = (idx & 3) * 8;
        __builtin_amdgcn_global_load_async_to_lds_b128(
            VPTR(&W[(size_t)(tile_n + rb) * K + k0 + cb]), VPTR(&Bs[rb][cb]), 0, 0);
      }
      __builtin_amdgcn_s_wait_asynccnt(0);
    }
#else
    for (int i = tid; i < 64 * 32; i += 256) {
      int r = i >> 5, c = i & 31, gr = tile_m + r, gc = k0 + c;
      As[r][c] = (gr < M && gc < K) ? A[(size_t)gr * K + gc] : (u16)0;
    }
    for (int i = tid; i < 128 * 32; i += 256) {
      int r = i >> 5, c = i & 31, gr = tile_n + r, gc = k0 + c;
      Bs[r][c] = (gr < N && gc < K) ? W[(size_t)gr * K + gc] : (u16)0;
    }
#endif
    __syncthreads();
    v16bf af[2], bw[2];
#pragma unroll
    for (int im = 0; im < 2; ++im)
      af[im] = load_frag_row(&As[0][0], wm * 32 + im * 16 + ln, 48, 0, hi);
#pragma unroll
    for (int in2 = 0; in2 < 2; ++in2)
      bw[in2] = load_frag_row(&Bs[0][0], wn * 32 + in2 * 16 + ln, 48, 0, hi);
#pragma unroll
    for (int im = 0; im < 2; ++im)
#pragma unroll
      for (int in2 = 0; in2 < 2; ++in2)
        acc[im][in2] = wmma_bf16(af[im], bw[in2], acc[im][in2]);
  }
#pragma unroll
  for (int im = 0; im < 2; ++im)
#pragma unroll
    for (int in2 = 0; in2 < 2; ++in2) {
      int col = tile_n + wn * 32 + in2 * 16 + ln;
      float bv = (bias != nullptr && col < N) ? bias[col] : 0.0f;
#pragma unroll
      for (int r = 0; r < 8; ++r) {
        int row = tile_m + wm * 32 + im * 16 + r + 8 * hi;
        if (row < M && col < N) {
          float v = acc[im][in2][r] + bv;
          if (outF) outF[(size_t)row * N + col] = v;
          if (outB) outB[(size_t)row * N + col] = f2bf(v);
        }
      }
    }
}

// ---- flash attention: one (b, h, 128-row q-block) per workgroup -----------
__global__ __launch_bounds__(256)
void flash_attn(const u16* __restrict__ qkv, u16* __restrict__ out) {
  constexpr int D = 96, BQ = 128, BK = 32, LD = 104;
  __shared__ u16 Qs[BQ][LD];
  __shared__ u16 Ks[BK][LD];
  __shared__ u16 Vs[BK][LD];
  __shared__ u16 Ps[8][16][36];  // per-wave P transpose staging (C -> A layout)

  const int qb = blockIdx.x, h = blockIdx.y, b = blockIdx.z;
  const int tid = threadIdx.x, lane = tid & 31, wave = tid >> 5;
  const int hi = lane >> 4, ln = lane & 15;
  const size_t rs = 3 * E_;
  const u16* qbase = qkv + (size_t)b * S_ * rs + h * D;
  const u16* kbase = qbase + E_;
  const u16* vbase = qbase + 2 * E_;

  for (int i = tid; i < BQ * D; i += 256) {
    int r = i / D, c = i % D;
    Qs[r][c] = qbase[(size_t)(qb * BQ + r) * rs + c];
  }
  v8f o[6] = {};
  float mrow[8], lrow[8];
#pragma unroll
  for (int r = 0; r < 8; ++r) { mrow[r] = -1e30f; lrow[r] = 0.0f; }
  const float scale = 0.1020620726f;  // 1/sqrt(96)

  for (int kt = 0; kt < S_; kt += BK) {
    __syncthreads();
    for (int i = tid; i < BK * D; i += 256) {
      int r = i / D, c = i % D;
      Ks[r][c] = kbase[(size_t)(kt + r) * rs + c];
      Vs[r][c] = vbase[(size_t)(kt + r) * rs + c];
    }
    __syncthreads();

    v8f s[2] = {};
#pragma unroll
    for (int kk = 0; kk < 3; ++kk) {
      v16bf aq = load_frag_row(&Qs[0][0], wave * 16 + ln, LD, kk * 32, hi);
#pragma unroll
      for (int nf = 0; nf < 2; ++nf) {
        v16bf bk = load_frag_row(&Ks[0][0], nf * 16 + ln, LD, kk * 32, hi);
        s[nf] = wmma_bf16(aq, bk, s[nf]);
      }
    }
    // online softmax: C-layout row = r + 8*hi, cols across 16-lane halves
#pragma unroll
    for (int r = 0; r < 8; ++r) {
      s[0][r] *= scale; s[1][r] *= scale;
      float v = fmaxf(s[0][r], s[1][r]);
#pragma unroll
      for (int off = 1; off < 16; off <<= 1) v = fmaxf(v, __shfl_xor(v, off, 32));
      float mnew = fmaxf(mrow[r], v);
      float alpha = __expf(mrow[r] - mnew);
      mrow[r] = mnew;
      float p0 = __expf(s[0][r] - mnew), p1 = __expf(s[1][r] - mnew);
      s[0][r] = p0; s[1][r] = p1;
      float ps = p0 + p1;
#pragma unroll
      for (int off = 1; off < 16; off <<= 1) ps += __shfl_xor(ps, off, 32);
      lrow[r] = lrow[r] * alpha + ps;
#pragma unroll
      for (int j = 0; j < 6; ++j) o[j][r] *= alpha;
    }
    // transpose P (C layout) -> A layout through per-wave LDS
#pragma unroll
    for (int r = 0; r < 8; ++r) {
      Ps[wave][r + 8 * hi][ln]      = f2bf(s[0][r]);
      Ps[wave][r + 8 * hi][16 + ln] = f2bf(s[1][r]);
    }
    asm volatile("s_wait_dscnt 0x0" ::: "memory");  // wave-local LDS RAW
    v16bf ap = load_frag_row(&Ps[wave][0][0], ln, 36, 0, hi);
#pragma unroll
    for (int nf = 0; nf < 6; ++nf) {
      v16bf bv;  // B(k,n) = V[k][n]: column reads, pairs span two LDS rows
#pragma unroll
      for (int i = 0; i < 8; ++i) {
        int kk = fragK(i, hi);
        bv[2 * i]     = asbf(Vs[kk][nf * 16 + ln]);
        bv[2 * i + 1] = asbf(Vs[kk + 1][nf * 16 + ln]);
      }
      o[nf] = wmma_bf16(ap, bv, o[nf]);
    }
  }
#pragma unroll
  for (int nf = 0; nf < 6; ++nf)
#pragma unroll
    for (int r = 0; r < 8; ++r) {
      int row = qb * BQ + wave * 16 + r + 8 * hi;
      int col = h * D + nf * 16 + ln;
      out[((size_t)b * S_ + row) * E_ + col] = f2bf(o[nf][r] / lrow[r]);
    }
}

// ---- GRU recurrence: blockIdx.x = direction; 32 waves; h resident in LDS --
// Each wave owns 3 gate-triples (r,z,n columns share a wave -> gates fuse in
// registers). Per step: 24 K-iters x 9 WMMA = 216 v_wmma per wave.
__global__ __launch_bounds__(1024)
void gru_pass(const u16* __restrict__ xw_f, const u16* __restrict__ xw_b,
              const u16* __restrict__ whh_f, const u16* __restrict__ whh_b,
              const float* __restrict__ bhh_f, const float* __restrict__ bhh_b,
              u16* __restrict__ hcat) {
  __shared__ u16 hs[32][776];  // 32 x 768 bf16 hidden state (+pad)
  const int dir = blockIdx.x;
  const u16*   xw  = dir ? xw_b  : xw_f;
  const u16*   whh = dir ? whh_b : whh_f;
  const float* bhh = dir ? bhh_b : bhh_f;
  const int tid = threadIdx.x, lane = tid & 31, wave = tid >> 5;
  const int hi = lane >> 4, ln = lane & 15;
  const int base_id = wave * 3;          // triples base_id..base_id+2
  const int mt = base_id / 48;           // same M-tile for all 3 (48 % 3 == 0)

  for (int i = tid; i < 32 * 776; i += 1024) (&hs[0][0])[i] = 0;
  __syncthreads();

  for (int t = 0; t < S_; ++t) {
    const int tt = dir ? (S_ - 1 - t) : t;
    v8f acc[3][3] = {};  // [triple][gate r,z,n]
    for (int k0 = 0; k0 < H_; k0 += 32) {
      v16bf ah = load_frag_row(&hs[0][0], mt * 16 + ln, 776, k0, hi);
#pragma unroll
      for (int j = 0; j < 3; ++j) {
        const int ntr = (base_id + j) % 48;
#pragma unroll
        for (int g = 0; g < 3; ++g) {
          const int nrow = g * H_ + ntr * 16 + ln;
          v16bf bw;  // Whh streamed from L2 each step (3.5MB bf16, resident)
#pragma unroll
          for (int i = 0; i < 8; ++i) {
            int kk = k0 + fragK(i, hi);
            bw[2 * i]     = asbf(whh[(size_t)nrow * H_ + kk]);
            bw[2 * i + 1] = asbf(whh[(size_t)nrow * H_ + kk + 1]);
          }
          acc[j][g] = wmma_bf16(ah, bw, acc[j][g]);
        }
      }
    }
    float hnew[3][8];
#pragma unroll
    for (int j = 0; j < 3; ++j) {
      const int c = ((base_id + j) % 48) * 16 + ln;
      const float bh_r = bhh[c], bh_z = bhh[H_ + c], bh_n = bhh[2 * H_ + c];
#pragma unroll
      for (int r = 0; r < 8; ++r) {
        const int m = mt * 16 + r + 8 * hi;
        const size_t xo = ((size_t)m * S_ + tt) * G3_ + c;
        float xr = bf2f(xw[xo]), xz = bf2f(xw[xo + H_]), xn = bf2f(xw[xo + 2 * H_]);
        float gr = acc[j][0][r] + bh_r;
        float gz = acc[j][1][r] + bh_z;
        float gn = acc[j][2][r] + bh_n;
        float rg = 1.0f / (1.0f + __expf(-(xr + gr)));
        float zg = 1.0f / (1.0f + __expf(-(xz + gz)));
        float nn = tanhf(xn + rg * gn);
        hnew[j][r] = (1.0f - zg) * nn + zg * bf2f(hs[m][c]);
      }
    }
    __syncthreads();  // all waves done reading h
#pragma unroll
    for (int j = 0; j < 3; ++j) {
      const int c = ((base_id + j) % 48) * 16 + ln;
#pragma unroll
      for (int r = 0; r < 8; ++r) {
        const int m = mt * 16 + r + 8 * hi;
        u16 hb = f2bf(hnew[j][r]);
        hs[m][c] = hb;
        hcat[((size_t)m * S_ + tt) * (2 * H_) + dir * H_ + c] = hb;
      }
    }
    __syncthreads();  // h updated before next step's reads
  }
}

// ---- tiny FC: em[BS][9] = h1cat @ fc_w^T + fc_b (N=9 < 16 -> VALU) --------
__global__ void fc_kernel(const u16* __restrict__ h1, const float* __restrict__ w,
                          const float* __restrict__ b, float* __restrict__ em) {
  int idx = blockIdx.x * 256 + threadIdx.x;
  if (idx >= BS_ * C_) return;
  int row = idx / C_, c = idx % C_;
  const u16* hp = h1 + (size_t)row * (2 * H_);
  const float* wp = w + (size_t)c * (2 * H_);
  float s = b[c];
  for (int k = 0; k < 2 * H_; ++k) s += bf2f(hp[k]) * wp[k];
  em[idx] = s;
}

// ---- CRF NLL + Viterbi: one lane per batch element ------------------------
__global__ void crf_kernel(const float* __restrict__ em, const int* __restrict__ labels,
                           const float* __restrict__ start, const float* __restrict__ endv,
                           const float* __restrict__ trans, float* __restrict__ out,
                           unsigned char* __restrict__ bp) {
  __shared__ float tr[C_ * C_], st[C_], en[C_];
  int tid = threadIdx.x;
  if (tid < C_ * C_) tr[tid] = trans[tid];
  if (tid < C_) { st[tid] = start[tid]; en[tid] = endv[tid]; }
  __syncthreads();
  if (tid < B_) {
    const int b = tid;
    const float* e = em + (size_t)b * S_ * C_;
    const int* lab = labels + (size_t)b * S_;
    float num = st[lab[0]] + e[lab[0]];
    for (int t = 1; t < S_; ++t) num += tr[lab[t - 1] * C_ + lab[t]] + e[t * C_ + lab[t]];
    num += en[lab[S_ - 1]];

    float alpha[C_], valpha[C_];
    for (int c = 0; c < C_; ++c) { alpha[c] = st[c] + e[c]; valpha[c] = alpha[c]; }
    for (int t = 1; t < S_; ++t) {
      float na[C_], nv[C_];
      for (int c = 0; c < C_; ++c) {
        float mx = -1e30f;
        for (int p = 0; p < C_; ++p) mx = fmaxf(mx, alpha[p] + tr[p * C_ + c]);
        float ss = 0.0f;
        for (int p = 0; p < C_; ++p) ss += __expf(alpha[p] + tr[p * C_ + c] - mx);
        na[c] = mx + __logf(ss) + e[t * C_ + c];
        float best = -1e30f; int bi = 0;
        for (int p = 0; p < C_; ++p) {
          float v = valpha[p] + tr[p * C_ + c];
          if (v > best) { best = v; bi = p; }
        }
        nv[c] = best + e[t * C_ + c];
        bp[((size_t)b * S_ + t) * C_ + c] = (unsigned char)bi;
      }
      for (int c = 0; c < C_; ++c) { alpha[c] = na[c]; valpha[c] = nv[c]; }
    }
    float mx = -1e30f;
    for (int c = 0; c < C_; ++c) mx = fmaxf(mx, alpha[c] + en[c]);
    float ss = 0.0f;
    for (int c = 0; c < C_; ++c) ss += __expf(alpha[c] + en[c] - mx);
    float nll = (mx + __logf(ss)) - num;
    for (int off = 16; off > 0; off >>= 1) nll += __shfl_down(nll, off, 32);
    if (tid == 0) out[0] = nll;

    int best = 0; float bv = -1e30f;
    for (int c = 0; c < C_; ++c) {
      float v = valpha[c] + en[c];
      if (v > bv) { bv = v; best = c; }
    }
    int* dec = (int*)(out + 1);  // int32 path bit-patterns after the loss
    dec[(size_t)b * S_ + (S_ - 1)] = best;
    int cur = best;
    for (int t = S_ - 1; t >= 1; --t) {
      cur = bp[((size_t)b * S_ + t) * C_ + cur];
      dec[(size_t)b * S_ + (t - 1)] = cur;
    }
  }
}

// ---------------------------------------------------------------------------
extern "C" void kernel_launch(void* const* d_in, const int* in_sizes, int n_in,
                              void* d_out, int out_size, void* d_ws, size_t ws_size,
                              hipStream_t stream) {
  (void)in_sizes; (void)n_in; (void)out_size; (void)ws_size;
  const float* emb   = (const float*)d_in[0];
  const float* inw   = (const float*)d_in[1];
  const float* inb   = (const float*)d_in[2];
  const float* outw  = (const float*)d_in[3];
  const float* outb  = (const float*)d_in[4];
  const float* fcw   = (const float*)d_in[5];
  const float* fcb   = (const float*)d_in[6];
  const float* cst   = (const float*)d_in[7];
  const float* cen   = (const float*)d_in[8];
  const float* ctr   = (const float*)d_in[9];
  const int*   lab   = (const int*)  d_in[10];
  const float* wih0f = (const float*)d_in[11];
  const float* whh0f = (const float*)d_in[12];
  const float* bih0f = (const float*)d_in[13];
  const float* bhh0f = (const float*)d_in[14];
  const float* wih0b = (const float*)d_in[15];
  const float* whh0b = (const float*)d_in[16];
  const float* bih0b = (const float*)d_in[17];
  const float* bhh0b = (const float*)d_in[18];
  const float* wih1f = (const float*)d_in[19];
  const float* whh1f = (const float*)d_in[20];
  const float* bih1f = (const float*)d_in[21];
  const float* bhh1f = (const float*)d_in[22];
  const float* wih1b = (const float*)d_in[23];
  const float* whh1b = (const float*)d_in[24];
  const float* bih1b = (const float*)d_in[25];
  const float* bhh1b = (const float*)d_in[26];

  size_t off = 0;
  auto alloc = [&](size_t bytes) -> void* {
    void* p = (char*)d_ws + off;
    off = (off + bytes + 255) & ~(size_t)255;
    return p;
  };
  u16* emb_bf = (u16*)alloc((size_t)BS_ * E_ * 2);
  u16* w_in   = (u16*)alloc((size_t)3 * E_ * E_ * 2);
  u16* w_out  = (u16*)alloc((size_t)E_ * E_ * 2);
  u16* w_ih0f = (u16*)alloc((size_t)G3_ * E_ * 2);
  u16* w_ih0b = (u16*)alloc((size_t)G3_ * E_ * 2);
  u16* w_hh0f = (u16*)alloc((size_t)G3_ * H_ * 2);
  u16* w_hh0b = (u16*)alloc((size_t)G3_ * H_ * 2);
  u16* w_ih1f = (u16*)alloc((size_t)G3_ * 2 * H_ * 2);
  u16* w_ih1b = (u16*)alloc((size_t)G3_ * 2 * H_ * 2);
  u16* w_hh1f = (u16*)alloc((size_t)G3_ * H_ * 2);
  u16* w_hh1b = (u16*)alloc((size_t)G3_ * H_ * 2);
  u16* qkvb   = (u16*)alloc((size_t)BS_ * 3 * E_ * 2);
  u16* attnb  = (u16*)alloc((size_t)BS_ * E_ * 2);
  u16* xb     = (u16*)alloc((size_t)BS_ * E_ * 2);
  u16* xw_f   = (u16*)alloc((size_t)BS_ * G3_ * 2);   // reused by layer 1
  u16* xw_b   = (u16*)alloc((size_t)BS_ * G3_ * 2);
  u16* h0cat  = (u16*)alloc((size_t)BS_ * 2 * H_ * 2);
  u16* h1cat  = (u16*)alloc((size_t)BS_ * 2 * H_ * 2);
  float* em   = (float*)alloc((size_t)BS_ * C_ * 4);
  unsigned char* bp = (unsigned char*)alloc((size_t)B_ * S_ * C_);

  auto cast = [&](const float* s, u16* d, size_t n) {
    cast_f32_bf16<<<(unsigned)((n + 255) / 256), 256, 0, stream>>>(s, d, n);
  };
  cast(emb,   emb_bf, (size_t)BS_ * E_);
  cast(inw,   w_in,   (size_t)3 * E_ * E_);
  cast(outw,  w_out,  (size_t)E_ * E_);
  cast(wih0f, w_ih0f, (size_t)G3_ * E_);
  cast(wih0b, w_ih0b, (size_t)G3_ * E_);
  cast(whh0f, w_hh0f, (size_t)G3_ * H_);
  cast(whh0b, w_hh0b, (size_t)G3_ * H_);
  cast(wih1f, w_ih1f, (size_t)G3_ * 2 * H_);
  cast(wih1b, w_ih1b, (size_t)G3_ * 2 * H_);
  cast(whh1f, w_hh1f, (size_t)G3_ * H_);
  cast(whh1b, w_hh1b, (size_t)G3_ * H_);

  dim3 g2304(G3_ / 128, BS_ / 64);   // 18 x 256
  dim3 g768(E_ / 128, BS_ / 64);     // 6 x 256

  // QKV projection
  wmma_gemm<<<g2304, 256, 0, stream>>>(emb_bf, w_in, inb, nullptr, qkvb,
                                       BS_, 3 * E_, E_);
  // flash attention
  flash_attn<<<dim3(S_ / 128, NH_, B_), 256, 0, stream>>>(qkvb, attnb);
  // output projection
  wmma_gemm<<<g768, 256, 0, stream>>>(attnb, w_out, outb, nullptr, xb,
                                      BS_, E_, E_);
  // GRU layer 0: input projections then recurrence (both directions)
  wmma_gemm<<<g2304, 256, 0, stream>>>(xb, w_ih0f, bih0f, nullptr, xw_f,
                                       BS_, G3_, E_);
  wmma_gemm<<<g2304, 256, 0, stream>>>(xb, w_ih0b, bih0b, nullptr, xw_b,
                                       BS_, G3_, E_);
  gru_pass<<<2, 1024, 0, stream>>>(xw_f, xw_b, w_hh0f, w_hh0b, bhh0f, bhh0b, h0cat);
  // GRU layer 1
  wmma_gemm<<<g2304, 256, 0, stream>>>(h0cat, w_ih1f, bih1f, nullptr, xw_f,
                                       BS_, G3_, 2 * H_);
  wmma_gemm<<<g2304, 256, 0, stream>>>(h0cat, w_ih1b, bih1b, nullptr, xw_b,
                                       BS_, G3_, 2 * H_);
  gru_pass<<<2, 1024, 0, stream>>>(xw_f, xw_b, w_hh1f, w_hh1b, bhh1f, bhh1b, h1cat);
  // FC + CRF
  fc_kernel<<<(BS_ * C_ + 255) / 256, 256, 0, stream>>>(h1cat, fcw, fcb, em);
  crf_kernel<<<1, 96, 0, stream>>>(em, lab, cst, cen, ctr, (float*)d_out, bp);
}
